// TGN_53472342835905
// MI455X (gfx1250) — matrix operand order repaired and checked by reading, not compiled, not run on hardware
//
#include <hip/hip_runtime.h>

typedef unsigned short u16;
typedef __bf16  v16bf __attribute__((ext_vector_type(16)));
typedef float   v8f   __attribute__((ext_vector_type(8)));
typedef u16     us8   __attribute__((ext_vector_type(8)));

#define D        128
#define NHEAD    2
#define DH       64
#define KNGH     16
#define BATCH    512
#define M3B      1536        // 3*B
#define M1       24576       // 3*B*K
#define M2       393216      // 3*B*K*K
#define N_NODES_ 131072
#define MSGD     100

// ---------- helpers ----------
__device__ __forceinline__ u16 f2bf(float f) {
    union { float f; unsigned int u; } c; c.f = f;
    unsigned int u = c.u;
    unsigned int r = (u + 0x7FFFu + ((u >> 16) & 1u)) >> 16;
    return (u16)r;
}
__device__ __forceinline__ float bf2f(u16 h) {
    union { unsigned int u; float f; } c; c.u = ((unsigned int)h) << 16;
    return c.f;
}

// One 16x16x32 bf16 WMMA step.
// A: 16x32 bf16 tile in LDS, row-major, stride aStride (elements).
//    Lane layout per ISA: lanes 0-15 -> M=lane, halves of K split by lane>>4;
//    elements 0..7 = K in [half*8, half*8+8), elements 8..15 = K in [16+half*8, ...).
// B: 32x16 bf16 tile in global (weights), row-major stride bStride; lane l holds column N=l&15, K in same order as A.
__device__ __forceinline__ v8f wmma_bf16_step(const u16* aBase, int aStride,
                                              const u16* bBase, int bStride, v8f c) {
    unsigned lane = threadIdx.x & 31u;
    unsigned half = lane >> 4;
    unsigned mr   = lane & 15u;
    union { us8 h[2]; v16bf v; } A;
    const u16* arow = aBase + mr * aStride + half * 8;
    A.h[0] = *(const us8*)(arow);
    A.h[1] = *(const us8*)(arow + 16);
    union { u16 h[16]; v16bf v; } B;
    const u16* brow = bBase + lane * bStride;
    __builtin_prefetch(brow + 32 * bStride, 0, 1);   // next K-step of weights
    const u16* bcol = bBase + mr;
#pragma unroll
    for (int s = 0; s < 8; s++) {
        B.h[s]     = bcol[(half * 8 + s) * bStride];
        B.h[8 + s] = bcol[(16 + half * 8 + s) * bStride];
    }
    return __builtin_amdgcn_wmma_f32_16x16x32_bf16(false, A.v, false, B.v,
                                                   (short)0, c, false, false);
}

// ---------- prep kernels ----------
__global__ void build_idx_kernel(const int* s, const int* t, const int* b,
                                 const float* cut, int* idx0, float* cut3) {
    int i = blockIdx.x * blockDim.x + threadIdx.x;
    if (i < M3B) {
        idx0[i] = (i < BATCH) ? s[i] : (i < 2 * BATCH) ? t[i - BATCH] : b[i - 2 * BATCH];
        cut3[i] = cut[i & (BATCH - 1)];
    }
}

__global__ void f32_to_bf16_kernel(const float* s, u16* d, int n) {
    int i = blockIdx.x * blockDim.x + threadIdx.x;
    if (i < n) d[i] = f2bf(s[i]);
}

// out[i,:] = bf16(node_feat[idx[i],:] + memory[idx[i],:])
__global__ void gather_feat_kernel(const int* idx, const float* nf, const float* mem,
                                   u16* out, int n) {
    long long e = (long long)blockIdx.x * blockDim.x + threadIdx.x;
    if (e < (long long)n * D) {
        int i = (int)(e >> 7), d = (int)(e & (D - 1));
        long long base = (long long)idx[i] * D + d;
        out[e] = f2bf(nf[base] + mem[base]);
    }
}

// ---------- fused temporal attention layer ----------
// One block = 16 source rows, 256 threads (8 wave32s), dynamic LDS.
__global__ void attn_layer_kernel(const u16* __restrict__ src_feat,   // M x D bf16
                                  const float* __restrict__ src_t,    // M
                                  const u16* __restrict__ ngh_feat,   // (M*K) x D bf16
                                  const int* __restrict__ ngh_e,      // M*K
                                  const float* __restrict__ ngh_t,    // M*K
                                  const float* __restrict__ edge_feat,// n_edges x D f32
                                  const float* __restrict__ tw, const float* __restrict__ tb,
                                  const u16* __restrict__ Wq,  // 256x128 bf16
                                  const u16* __restrict__ Wk,  // 384x128 bf16
                                  const u16* __restrict__ Wv,  // 384x128 bf16
                                  const u16* __restrict__ Wo,  // 128x128 bf16
                                  const u16* __restrict__ mw1, const float* __restrict__ mb1,
                                  const u16* __restrict__ mw2, const float* __restrict__ mb2,
                                  u16* __restrict__ out_bf16,  // M x D
                                  int M) {
    extern __shared__ char smem_raw[];
    u16*   s_qin  = (u16*)  (smem_raw);             // 16x256 bf16  (8192B)
    float* s_q    = (float*)(smem_raw + 8192);      // 16x128 f32   (8192B)
    u16*   s_kin  = (u16*)  (smem_raw + 16384);     // 64x384 bf16  (49152B)
    float* s_k    = (float*)(smem_raw + 65536);     // 64x128 f32   (32768B)
    float* s_v    = (float*)(smem_raw + 98304);     // 64x128 f32   (32768B)
    float* s_attn = (float*)(smem_raw + 131072);    // 4x2x16 f32   (512B)
    u16*   s_obf  = (u16*)  (smem_raw + 131584);    // 16x128 bf16  (4096B)
    u16*   s_cat  = (u16*)  (smem_raw + 135680);    // 16x256 bf16  (8192B)
    u16*   s_h    = (u16*)  (smem_raw + 143872);    // 16x128 bf16  (4096B)

    const int tid  = threadIdx.x;
    const int wave = tid >> 5;
    const int lane = tid & 31;
    const int m0   = blockIdx.x * 16;

    // ---- q_in = [src_feat | cos(0*w + b)] ----
    for (int e = tid; e < 16 * 256; e += 256) {
        int r = e >> 8, col = e & 255;
        u16 v;
        if (col < D) v = src_feat[(long long)(m0 + r) * D + col];
        else         v = f2bf(cosf(tb[col - D]));
        s_qin[r * 256 + col] = v;
    }
    __syncthreads();

    // ---- q = q_in @ Wq : 1 M-tile x 8 N-tiles, 8 K-steps; wave w -> N-tile w ----
    {
        v8f acc = {};
        for (int ks = 0; ks < 8; ks++)
            acc = wmma_bf16_step(&s_qin[ks * 32], 256, Wq + ks * 32 * D + wave * 16, D, acc);
        int n = (lane & 15) + wave * 16;
#pragma unroll
        for (int i = 0; i < 8; i++) {
            int m = i + 8 * (lane >> 4);
            s_q[m * D + n] = acc[i];
        }
    }
    __syncthreads();

    // ---- neighbor chunks: 4 source rows x 16 neighbors = 64 k_in rows each ----
    for (int c = 0; c < 4; c++) {
        // (1) async DMA: ngh_feat rows -> s_kin[:, 0:128]   (64 rows x 256B = 1024 x 16B)
        for (int seg = tid; seg < 1024; seg += 256) {
            int r  = seg >> 4;           // k_in row 0..63
            int sc = seg & 15;           // 16B segment within the 256B feature row
            int m  = m0 + c * 4 + (r >> 4);
            int kn = r & 15;
            long long gk = (long long)m * KNGH + kn;
            const u16* gptr = ngh_feat + gk * D + sc * 8;
            unsigned lds_lo = (unsigned)(size_t)(&s_kin[r * 384 + sc * 8]);
            unsigned long long ga = (unsigned long long)(size_t)gptr;
            asm volatile("global_load_async_to_lds_b128 %0, %1, off"
                         :: "v"(lds_lo), "v"(ga) : "memory");
        }
        // (2) VALU fill: edge features + time encoding -> s_kin[:, 128:384]
        for (int e = tid; e < 64 * 256; e += 256) {
            int r   = e >> 8;
            int col = D + (e & 255);
            int m  = m0 + c * 4 + (r >> 4);
            int kn = r & 15;
            long long gk = (long long)m * KNGH + kn;
            u16 v;
            if (col < 2 * D) {
                long long eidx = (long long)ngh_e[gk];
                v = f2bf(edge_feat[eidx * D + (col - D)]);
            } else {
                int d = col - 2 * D;
                float dt = src_t[m] - ngh_t[gk];
                v = f2bf(cosf(dt * tw[d] + tb[d]));
            }
            s_kin[r * 384 + col] = v;
        }
        asm volatile("s_wait_asynccnt 0x0" ::: "memory");
        __syncthreads();

        // k = k_in@Wk, v = k_in@Wv : 64 WMMA jobs (4 Mt x 8 Nt x {k,v}), 8 per wave
        for (int j = 0; j < 8; j++) {
            int job = wave * 8 + j;
            int isv = job >> 5;
            int t2  = job & 31;
            int mt  = t2 >> 3, nt = t2 & 7;
            const u16* W = isv ? Wv : Wk;
            v8f acc = {};
            for (int ks = 0; ks < 12; ks++)
                acc = wmma_bf16_step(&s_kin[(mt * 16) * 384 + ks * 32], 384,
                                     W + ks * 32 * D + nt * 16, D, acc);
            float* dst = isv ? s_v : s_k;
            int n = (lane & 15) + nt * 16;
#pragma unroll
            for (int i = 0; i < 8; i++) {
                int m = mt * 16 + i + 8 * (lane >> 4);
                dst[m * D + n] = acc[i];
            }
        }
        __syncthreads();

        // scores + masked softmax over K=16 (half-wave shuffle groups)
        if (tid < 128) {
            int rr = tid >> 5;           // wave index = source row within chunk
            int h  = (lane >> 4) & 1;    // head
            int kn = lane & 15;
            int m  = m0 + c * 4 + rr;
            const float* qp = &s_q[(c * 4 + rr) * D + h * DH];
            const float* kp = &s_k[(rr * KNGH + kn) * D + h * DH];
            float s = 0.f;
            for (int d = 0; d < DH; d++) s += qp[d] * kp[d];
            s *= 0.125f;                 // 1/sqrt(64)
            if (ngh_e[(long long)m * KNGH + kn] == 0) s = -1.0e9f;
            float mx = s;
            for (int o = 8; o >= 1; o >>= 1) mx = fmaxf(mx, __shfl_xor(mx, o, 32));
            float ex = expf(s - mx);
            float sm = ex;
            for (int o = 8; o >= 1; o >>= 1) sm += __shfl_xor(sm, o, 32);
            s_attn[(rr * 2 + h) * 16 + kn] = ex / sm;
        }
        __syncthreads();

        // out[rr,d] = sum_kn attn * v
        for (int e = tid; e < 512; e += 256) {
            int rr = e >> 7, d = e & (D - 1);
            int h = d >> 6;
            float acc2 = 0.f;
            for (int kn = 0; kn < KNGH; kn++)
                acc2 += s_attn[(rr * 2 + h) * 16 + kn] * s_v[(rr * KNGH + kn) * D + d];
            s_obf[(c * 4 + rr) * D + d] = f2bf(acc2);
        }
        __syncthreads();
    }

    // ---- out2 = out @ Wo -> s_cat[:, 0:128] ----
    {
        v8f acc = {};
        for (int ks = 0; ks < 4; ks++)
            acc = wmma_bf16_step(&s_obf[ks * 32], D, Wo + ks * 32 * D + wave * 16, D, acc);
        int n = (lane & 15) + wave * 16;
#pragma unroll
        for (int i = 0; i < 8; i++) {
            int m = i + 8 * (lane >> 4);
            s_cat[m * 256 + n] = f2bf(acc[i]);
        }
    }
    // s_cat[:, 128:256] = src_feat
    for (int e = tid; e < 16 * D; e += 256) {
        int r = e >> 7, d = e & (D - 1);
        s_cat[r * 256 + D + d] = src_feat[(long long)(m0 + r) * D + d];
    }
    __syncthreads();

    // ---- h = relu(cat @ mw1 + mb1) ----
    {
        v8f acc = {};
        for (int ks = 0; ks < 8; ks++)
            acc = wmma_bf16_step(&s_cat[ks * 32], 256, mw1 + ks * 32 * D + wave * 16, D, acc);
        int n = (lane & 15) + wave * 16;
#pragma unroll
        for (int i = 0; i < 8; i++) {
            int m = i + 8 * (lane >> 4);
            s_h[m * D + n] = f2bf(fmaxf(acc[i] + mb1[n], 0.f));
        }
    }
    __syncthreads();

    // ---- final = h @ mw2 + mb2 -> global bf16 ----
    {
        v8f acc = {};
        for (int ks = 0; ks < 4; ks++)
            acc = wmma_bf16_step(&s_h[ks * 32], D, mw2 + ks * 32 * D + wave * 16, D, acc);
        int n = (lane & 15) + wave * 16;
#pragma unroll
        for (int i = 0; i < 8; i++) {
            int m = i + 8 * (lane >> 4);
            out_bf16[(long long)(m0 + m) * D + n] = f2bf(acc[i] + mb2[n]);
        }
    }
}

// ---------- affinity head: score = relu(x@w1+b1)@w2 + b2 ----------
__global__ void affinity_kernel(const u16* emb, const float* w1, const float* b1,
                                const float* w2, const float* b2, float* out) {
    int r = blockIdx.x;          // 0..1023
    int j = threadIdx.x;         // 128
    __shared__ float xrow[256];
    __shared__ float red[128];
    int rr = r & (BATCH - 1);
    const u16* a = emb + (long long)rr * D;
    const u16* bb = emb + (long long)((r < BATCH) ? (BATCH + rr) : (2 * BATCH + rr)) * D;
    xrow[j]       = bf2f(a[j]);
    xrow[D + j]   = bf2f(bb[j]);
    __syncthreads();
    float h = b1[j];
    for (int i = 0; i < 2 * D; i++) h += xrow[i] * w1[i * D + j];
    red[j] = fmaxf(h, 0.f) * w2[j];
    __syncthreads();
    for (int s2 = 64; s2 > 0; s2 >>= 1) {
        if (j < s2) red[j] += red[j + s2];
        __syncthreads();
    }
    if (j == 0) out[r] = red[0] + b2[0];
}

// ---------- message MLP: msg = relu(raw@w1+b1)@w2 + b2 ----------
__global__ void message_kernel(const u16* emb, const int* e_idx, const int* src_idx,
                               const int* tgt_idx, const float* cut, const float* last_update,
                               const float* edge_feat, const float* tw, const float* tb,
                               const float* w1, const float* b1,
                               const float* w2, const float* b2, float* msg) {
    int r = blockIdx.x;          // 0..1023
    int t = threadIdx.x;         // 256
    __shared__ float raw[512];
    __shared__ float h1[256];
    int rr = r & (BATCH - 1);
    bool second = (r >= BATCH);
    const u16* eA = emb + (long long)(second ? (BATCH + rr) : rr) * D;
    const u16* eB = emb + (long long)(second ? rr : (BATCH + rr)) * D;
    int node = second ? tgt_idx[rr] : src_idx[rr];
    long long ei = (long long)e_idx[rr];
    for (int i = t; i < 512; i += 256) {
        float v;
        if (i < D)            v = bf2f(eA[i]);
        else if (i < 2 * D)   v = bf2f(eB[i - D]);
        else if (i < 3 * D)   v = edge_feat[ei * D + (i - 2 * D)];
        else {
            int d = i - 3 * D;
            v = cosf((cut[rr] - last_update[node]) * tw[d] + tb[d]);
        }
        raw[i] = v;
    }
    __syncthreads();
    {
        float a = b1[t];
        for (int i = 0; i < 512; i++) a += raw[i] * w1[i * 256 + t];
        h1[t] = fmaxf(a, 0.f);
    }
    __syncthreads();
    if (t < MSGD) {
        float a = b2[t];
        for (int i = 0; i < 256; i++) a += h1[i] * w2[i * MSGD + t];
        msg[r * MSGD + t] = a;
    }
}

// ---------- GRU memory update ----------
__global__ void gru_kernel(const float* msg, const float* memory, const int* src_idx,
                           const int* tgt_idx, const float* wih, const float* whh,
                           const float* bih, const float* bhh, float* hnew) {
    int r = blockIdx.x;          // 0..1023
    int j = threadIdx.x;         // 128
    int node = (r < BATCH) ? src_idx[r] : tgt_idx[r - BATCH];
    __shared__ float hprev[D];
    __shared__ float m_[MSGD];
    hprev[j] = memory[(long long)node * D + j];
    if (j < MSGD) m_[j] = msg[r * MSGD + j];
    __syncthreads();
    float gi[3], gh[3];
    for (int g = 0; g < 3; g++) {
        int row = g * D + j;
        float a = bih[row];
        for (int i = 0; i < MSGD; i++) a += m_[i] * wih[row * MSGD + i];
        gi[g] = a;
        float cacc = bhh[row];
        for (int i = 0; i < D; i++) cacc += hprev[i] * whh[row * D + i];
        gh[g] = cacc;
    }
    float rg = 1.f / (1.f + expf(-(gi[0] + gh[0])));
    float z  = 1.f / (1.f + expf(-(gi[1] + gh[1])));
    float n  = tanhf(gi[2] + rg * gh[2]);
    hnew[r * D + j] = (1.f - z) * n + z * hprev[j];
}

__global__ void scatter_kernel(const float* hnew, const int* src_idx, const int* tgt_idx,
                               float* mem_out) {
    int r = blockIdx.x;
    int j = threadIdx.x;
    int node = (r < BATCH) ? src_idx[r] : tgt_idx[r - BATCH];
    mem_out[(long long)node * D + j] = hnew[r * D + j];
}

// ---------- launch ----------
extern "C" void kernel_launch(void* const* d_in, const int* in_sizes, int n_in,
                              void* d_out, int out_size, void* d_ws, size_t ws_size,
                              hipStream_t stream) {
    const int*   src_idx   = (const int*)  d_in[0];
    const int*   tgt_idx   = (const int*)  d_in[1];
    const int*   bgd_idx   = (const int*)  d_in[2];
    const int*   e_idx     = (const int*)  d_in[3];
    const int*   ngh1      = (const int*)  d_in[4];
    const int*   eidx1     = (const int*)  d_in[5];
    const int*   ngh2      = (const int*)  d_in[6];
    const int*   eidx2     = (const int*)  d_in[7];
    const float* cut_time  = (const float*)d_in[8];
    const float* ts1       = (const float*)d_in[9];
    const float* ts2       = (const float*)d_in[10];
    const float* node_feat = (const float*)d_in[11];
    const float* edge_feat = (const float*)d_in[12];
    const float* memory    = (const float*)d_in[13];
    const float* last_upd  = (const float*)d_in[14];
    const float* time_w    = (const float*)d_in[15];
    const float* time_b    = (const float*)d_in[16];
    const float* Wq        = (const float*)d_in[17];
    const float* Wk        = (const float*)d_in[18];
    const float* Wv        = (const float*)d_in[19];
    const float* Wo        = (const float*)d_in[20];
    const float* mw1       = (const float*)d_in[21];
    const float* mb1       = (const float*)d_in[22];
    const float* mw2       = (const float*)d_in[23];
    const float* mb2       = (const float*)d_in[24];
    const float* msg_w1    = (const float*)d_in[25];
    const float* msg_b1    = (const float*)d_in[26];
    const float* msg_w2    = (const float*)d_in[27];
    const float* msg_b2    = (const float*)d_in[28];
    const float* gru_wih   = (const float*)d_in[29];
    const float* gru_whh   = (const float*)d_in[30];
    const float* gru_bih   = (const float*)d_in[31];
    const float* gru_bhh   = (const float*)d_in[32];
    const float* aff_w1    = (const float*)d_in[33];
    const float* aff_b1    = (const float*)d_in[34];
    const float* aff_w2    = (const float*)d_in[35];
    const float* aff_b2    = (const float*)d_in[36];

    char* ws = (char*)d_ws;
    size_t off = 0;
    auto alloc = [&](size_t bytes) -> void* {
        void* p = ws + off;
        off = (off + bytes + 255) & ~(size_t)255;
        return p;
    };
    int*   idx0   = (int*)  alloc((size_t)M3B * 4);
    float* cut3   = (float*)alloc((size_t)M3B * 4);
    u16*   f0     = (u16*)  alloc((size_t)M3B * D * 2);
    u16*   f1     = (u16*)  alloc((size_t)M1 * D * 2);
    u16*   f2     = (u16*)  alloc((size_t)M2 * D * 2);
    u16*   emb1   = (u16*)  alloc((size_t)M1 * D * 2);
    u16*   emb0   = (u16*)  alloc((size_t)M3B * D * 2);
    u16*   embf   = (u16*)  alloc((size_t)M3B * D * 2);
    u16*   Wq_bf  = (u16*)  alloc((size_t)2 * 256 * D * 2);
    u16*   Wk_bf  = (u16*)  alloc((size_t)2 * 384 * D * 2);
    u16*   Wv_bf  = (u16*)  alloc((size_t)2 * 384 * D * 2);
    u16*   Wo_bf  = (u16*)  alloc((size_t)2 * D * D * 2);
    u16*   mw1_bf = (u16*)  alloc((size_t)2 * 256 * D * 2);
    u16*   mw2_bf = (u16*)  alloc((size_t)2 * D * D * 2);
    float* msg    = (float*)alloc((size_t)1024 * MSGD * 4);
    float* hnew   = (float*)alloc((size_t)1024 * D * 4);

    build_idx_kernel<<<(M3B + 255) / 256, 256, 0, stream>>>(src_idx, tgt_idx, bgd_idx,
                                                            cut_time, idx0, cut3);

    auto cvt = [&](const float* s, u16* d, int n) {
        f32_to_bf16_kernel<<<(n + 255) / 256, 256, 0, stream>>>(s, d, n);
    };
    cvt(Wq,  Wq_bf,  2 * 256 * D);
    cvt(Wk,  Wk_bf,  2 * 384 * D);
    cvt(Wv,  Wv_bf,  2 * 384 * D);
    cvt(Wo,  Wo_bf,  2 * D * D);
    cvt(mw1, mw1_bf, 2 * 256 * D);
    cvt(mw2, mw2_bf, 2 * D * D);

    gather_feat_kernel<<<(M3B * D) / 256, 256, 0, stream>>>(idx0, node_feat, memory, f0, M3B);
    gather_feat_kernel<<<(M1  * D) / 256, 256, 0, stream>>>(ngh1, node_feat, memory, f1, M1);
    gather_feat_kernel<<<(M2  * D) / 256, 256, 0, stream>>>(ngh2, node_feat, memory, f2, M2);

    const size_t SMEM = 147968;
    // layer-0 on second-hop neighborhood: emb1
    attn_layer_kernel<<<M1 / 16, 256, SMEM, stream>>>(
        f1, ts1, f2, eidx2, ts2, edge_feat, time_w, time_b,
        Wq_bf, Wk_bf, Wv_bf, Wo_bf, mw1_bf, mb1, mw2_bf, mb2, emb1, M1);
    // layer-0 on first-hop neighborhood: emb0
    attn_layer_kernel<<<M3B / 16, 256, SMEM, stream>>>(
        f0, cut3, f1, eidx1, ts1, edge_feat, time_w, time_b,
        Wq_bf, Wk_bf, Wv_bf, Wo_bf, mw1_bf, mb1, mw2_bf, mb2, emb0, M3B);
    // layer-1: emb
    attn_layer_kernel<<<M3B / 16, 256, SMEM, stream>>>(
        emb0, cut3, emb1, eidx1, ts1, edge_feat, time_w, time_b,
        Wq_bf + 256 * D, Wk_bf + 384 * D, Wv_bf + 384 * D, Wo_bf + D * D,
        mw1_bf + 256 * D, mb1 + D, mw2_bf + D * D, mb2 + D, embf, M3B);

    float* out_f = (float*)d_out;
    affinity_kernel<<<1024, 128, 0, stream>>>(embf, aff_w1, aff_b1, aff_w2, aff_b2, out_f);

    message_kernel<<<1024, 256, 0, stream>>>(embf, e_idx, src_idx, tgt_idx, cut_time,
                                             last_upd, edge_feat, time_w, time_b,
                                             msg_w1, msg_b1, msg_w2, msg_b2, msg);

    hipMemcpyAsync(out_f + 1024, memory, (size_t)N_NODES_ * D * sizeof(float),
                   hipMemcpyDeviceToDevice, stream);

    gru_kernel<<<1024, 128, 0, stream>>>(msg, memory, src_idx, tgt_idx,
                                         gru_wih, gru_whh, gru_bih, gru_bhh, hnew);
    scatter_kernel<<<1024, 128, 0, stream>>>(hnew, src_idx, tgt_idx, out_f + 1024);
}
